// Model_68289980006983
// MI455X (gfx1250) — matrix operand order repaired and checked by reading, not compiled, not run on hardware
//
#include <hip/hip_runtime.h>
#include <cstdint>

// Problem constants
#define BB 64
#define TT 512
#define FF 64
#define UU 128
#define NG 512              // 4*U
#define BT (BB*TT)          // 32768

typedef __bf16 bf16;
typedef __attribute__((ext_vector_type(16))) __bf16 v16bf;
typedef __attribute__((ext_vector_type(8)))  float  v8f;

// ---------- scalar helpers ----------
__device__ __forceinline__ float bf2f(bf16 b) {
    unsigned short u; __builtin_memcpy(&u, &b, 2);
    unsigned x = ((unsigned)u) << 16;
    float f; __builtin_memcpy(&f, &x, 4);
    return f;
}
__device__ __forceinline__ bf16 f2bf(float f) {
    unsigned u; __builtin_memcpy(&u, &f, 4);
    unsigned r = u + 0x7FFFu + ((u >> 16) & 1u);   // RNE
    unsigned short h = (unsigned short)(r >> 16);
    bf16 o; __builtin_memcpy(&o, &h, 2);
    return o;
}
__device__ __forceinline__ unsigned packf2(float a, float b) {
    bf16 x = f2bf(a), y = f2bf(b);
    unsigned short xs, ys;
    __builtin_memcpy(&xs, &x, 2); __builtin_memcpy(&ys, &y, 2);
    return (unsigned)xs | ((unsigned)ys << 16);
}
__device__ __forceinline__ float bflo(unsigned p) {
    unsigned x = p << 16; float f; __builtin_memcpy(&f, &x, 4); return f;
}
__device__ __forceinline__ float bfhi(unsigned p) {
    unsigned x = p & 0xFFFF0000u; float f; __builtin_memcpy(&f, &x, 4); return f;
}
// fast sigmoid/tanh on the sequential critical path
__device__ __forceinline__ float fsigmoid(float x) {
    return __builtin_amdgcn_rcpf(1.0f + __expf(-x));
}
__device__ __forceinline__ float ftanh(float x) {
    return 1.0f - 2.0f * __builtin_amdgcn_rcpf(1.0f + __expf(2.0f * x));
}

// ---------- WMMA fragment loaders (CDNA5 16x16x32 bf16, ISA 7.12.2) ----------
// A-matrix 16x32 row-major: per lane two contiguous 16B runs.
__device__ __forceinline__ v16bf load_a_frag(const bf16* A, int row0, int col0, int lda) {
    int lane = threadIdx.x & 31;
    int half = lane >> 4;
    const uint4* p = (const uint4*)(A + (size_t)(row0 + (lane & 15)) * lda + col0);
    union { v16bf v; uint4 q[2]; } r;
    r.q[0] = p[half];
    r.q[1] = p[2 + half];
    return r.v;
}
// B fragment from pre-packed layout: ((kt*ntn + tn)*32 + lane)*8 dwords, contiguous 32B.
__device__ __forceinline__ v16bf load_b_packed(const unsigned* P, int kt, int tn, int ntn) {
    int lane = threadIdx.x & 31;
    const uint4* p = (const uint4*)(P + (((size_t)(kt * ntn + tn) * 32 + lane) << 3));
    union { v16bf v; uint4 q[2]; } r;
    r.q[0] = p[0];
    r.q[1] = p[1];
    return r.v;
}
__device__ __forceinline__ v8f wmma_bf16(v16bf a, v16bf b, v8f c) {
    return __builtin_amdgcn_wmma_f32_16x16x32_bf16(false, a, false, b, (short)0, c, false, false);
}

// ---------- kernels ----------
__global__ void cvt_f32_bf16(const float* __restrict__ src, bf16* __restrict__ dst, int n) {
    int i = blockIdx.x * blockDim.x + threadIdx.x;
    if (i < n) dst[i] = f2bf(src[i]);
}

// fp32 weight (K x N row-major) -> bf16 WMMA B-fragment layout.
__global__ void pack_weights(const float* __restrict__ src, unsigned* __restrict__ dst,
                             int K, int N) {
    int idx = blockIdx.x * blockDim.x + threadIdx.x;
    int ntn = N >> 4;
    int total = (K >> 5) * ntn * 256;
    if (idx >= total) return;
    int i    = idx & 7;
    int lane = (idx >> 3) & 31;
    int tile = idx >> 8;
    int tn = tile % ntn, kt = tile / ntn;
    int n = tn * 16 + (lane & 15);
    int k = kt * 32 + (lane >> 4) * 8 + (i >> 2) * 16 + 2 * (i & 3);
    dst[idx] = packf2(src[(size_t)k * N + n], src[(size_t)(k + 1) * N + n]);
}

__global__ void mask_kernel(const float* __restrict__ x, float* __restrict__ mask) {
    int idx = blockIdx.x * blockDim.x + threadIdx.x;   // over B*T
    if (idx >= BT) return;
    const float* row = x + (size_t)idx * FF;
    float any = 0.0f;
#pragma unroll 8
    for (int i = 0; i < FF; ++i) any = (row[i] != 0.0f) ? 1.0f : any;
    mask[idx] = any;
}

// C = A[M,K]bf16 @ Bpacked[K,N] + bias[N], written TIME-MAJOR: row m=b*TT+t -> row t*BB+b.
// 32x32 output per wave (2x2 WMMA tiles) for 2x fragment reuse.
__global__ void gemm_bf16(const bf16* __restrict__ A, const unsigned* __restrict__ Bp,
                          const float* __restrict__ bias, float* __restrict__ C,
                          int M, int N, int K) {
    int wave = blockIdx.x * (blockDim.x >> 5) + (threadIdx.x >> 5);
    int ntp = N >> 5;                  // tn-pairs
    int total = (M >> 5) * ntp;
    if (wave >= total) return;         // wave-uniform
    int tm0 = (wave / ntp) * 2;
    int tn0 = (wave % ntp) * 2;
    int ntn = N >> 4;
    v8f acc00 = {}, acc01 = {}, acc10 = {}, acc11 = {};
    for (int kt = 0; kt < (K >> 5); ++kt) {
        v16bf a0 = load_a_frag(A, tm0 * 16, kt * 32, K);
        v16bf a1 = load_a_frag(A, tm0 * 16 + 16, kt * 32, K);
        v16bf b0 = load_b_packed(Bp, kt, tn0, ntn);
        v16bf b1 = load_b_packed(Bp, kt, tn0 + 1, ntn);
        acc00 = wmma_bf16(a0, b0, acc00);
        acc01 = wmma_bf16(a0, b1, acc01);
        acc10 = wmma_bf16(a1, b0, acc10);
        acc11 = wmma_bf16(a1, b1, acc11);
    }
    int lane = threadIdx.x & 31;
    int half = lane >> 4;
    int ln = lane & 15;
    v8f accs[4] = {acc00, acc01, acc10, acc11};
#pragma unroll
    for (int q = 0; q < 4; ++q) {
        int tm = tm0 + (q >> 1);
        int tn = tn0 + (q & 1);
        int n = tn * 16 + ln;
        float bv = bias[n];
#pragma unroll
        for (int r = 0; r < 8; ++r) {
            int m = tm * 16 + r + half * 8;
            int b = m >> 9;            // / TT
            int t = m & (TT - 1);
            C[((size_t)t * BB + b) * NG + n] = accs[q][r] + bv;
        }
    }
}

// One block per direction; 512 threads (16 waves) on one WGP.
// LDS: Wh-packed 128K + z 128K + c 32K + h 16K = 304 KB (CDNA5 320K WGP LDS).
// zx layout is time-major: zx[t][b][n].
// Per step, wave w owns N-tiles {2w, 2w+1} x all 4 M-tiles: per k-step
// 4 A frags + 2 B frags (12 ds_load_b128) feed 8 WMMAs.
__global__ void lstm_scan(const float* __restrict__ zxF, const float* __restrict__ zxB,
                          const unsigned* __restrict__ WhF, const unsigned* __restrict__ WhB,
                          const float* __restrict__ mask, bf16* __restrict__ h_out,
                          int store_all) {
    extern __shared__ char smem[];
    unsigned* sWh = (unsigned*)smem;             // 131072 B (packed B fragments, ntn=32)
    float*    sZ  = (float*)(smem + 131072);     // 131072 B
    float*    sC  = (float*)(smem + 262144);     //  32768 B
    bf16*     sH  = (bf16*)(smem + 294912);      //  16384 B

    const int dir = blockIdx.x;                  // 0 = forward, 1 = backward
    const float*    zx = dir ? zxB : zxF;
    const unsigned* Wh = dir ? WhB : WhF;

    const int tid  = threadIdx.x;
    const int wave = tid >> 5;
    const int lane = tid & 31;
    const int half = lane >> 4;
    const int ln   = lane & 15;
    const int tn0  = wave * 2;       // this wave's two N-tiles

    // Load packed Wh (128 KB) into LDS: 8192 uint4
    {
        const uint4* src = (const uint4*)Wh;
        uint4* dst = (uint4*)sWh;
        for (int i = tid; i < 8192; i += 512) dst[i] = src[i];
    }
    // Init h, c
    for (int i = tid; i < (BB * UU) / 2; i += 512) ((unsigned*)sH)[i] = 0u;
    for (int i = tid; i < BB * UU; i += 512) sC[i] = 0.0f;

    for (int step = 0; step < TT; ++step) {
        const int t = dir ? (TT - 1 - step) : step;
        const int t_next = dir ? (t - 1) : (t + 1);
        __syncthreads();

        const float* zxt = zx + (size_t)t * (BB * NG);
        const float* zxn = (step + 1 < TT) ? (zx + (size_t)t_next * (BB * NG)) : zxt;

        // z = h @ Wh + zx   (M=64, N=512, K=128)
        v8f acc[8] = {};   // [tm][p]  p in {0,1}
#pragma unroll
        for (int kt = 0; kt < 4; ++kt) {
            v16bf b0 = load_b_packed(sWh, kt, tn0, 32);
            v16bf b1 = load_b_packed(sWh, kt, tn0 + 1, 32);
#pragma unroll
            for (int tm = 0; tm < 4; ++tm) {
                v16bf a = load_a_frag(sH, tm * 16, kt * 32, UU);
                acc[tm * 2 + 0] = wmma_bf16(a, b0, acc[tm * 2 + 0]);
                acc[tm * 2 + 1] = wmma_bf16(a, b1, acc[tm * 2 + 1]);
            }
        }
#pragma unroll
        for (int q = 0; q < 8; ++q) {
            int tm = q >> 1;
            int n = (tn0 + (q & 1)) * 16 + ln;
#pragma unroll
            for (int r = 0; r < 8; ++r) {
                int bidx = tm * 16 + r + half * 8;
                sZ[bidx * NG + n] = acc[q][r] + zxt[(size_t)bidx * NG + n];
            }
            // overlap next step's zx stream with this step's gate math
            __builtin_prefetch(&zxn[(size_t)(tm * 16) * NG + n], 0, 0);
        }
        __syncthreads();

        // gates + state update: two adjacent cells (same b, u even) per iteration
        for (int base = tid * 2; base < BB * UU; base += 1024) {
            int b = base >> 7, u = base & 127;
            float m = mask[(size_t)b * TT + t];
            unsigned hold = ((const unsigned*)sH)[base >> 1];

            float zi0 = sZ[b * NG + u],       zi1 = sZ[b * NG + u + 1];
            float zf0 = sZ[b * NG + 128 + u], zf1 = sZ[b * NG + 129 + u];
            float zg0 = sZ[b * NG + 256 + u], zg1 = sZ[b * NG + 257 + u];
            float zo0 = sZ[b * NG + 384 + u], zo1 = sZ[b * NG + 385 + u];

            float c0 = sC[base], c1 = sC[base + 1];
            float cn0 = fsigmoid(zf0) * c0 + fsigmoid(zi0) * ftanh(zg0);
            float cn1 = fsigmoid(zf1) * c1 + fsigmoid(zi1) * ftanh(zg1);
            float hn0 = fsigmoid(zo0) * ftanh(cn0);
            float hn1 = fsigmoid(zo1) * ftanh(cn1);

            bool upd = (m != 0.0f);
            float hs0 = upd ? hn0 : bflo(hold);
            float hs1 = upd ? hn1 : bfhi(hold);
            sC[base]     = upd ? cn0 : c0;
            sC[base + 1] = upd ? cn1 : c1;
            unsigned hp = packf2(hs0, hs1);
            ((unsigned*)sH)[base >> 1] = hp;
            if (store_all)
                ((unsigned*)h_out)[(((size_t)b * TT + t) * (2 * UU) + dir * UU + u) >> 1] = hp;
        }
    }
    __syncthreads();
    if (!store_all) {
        for (int base = tid * 2; base < BB * UU; base += 1024) {
            int b = base >> 7, u = base & 127;
            ((unsigned*)h_out)[((size_t)b * (2 * UU) + dir * UU + u) >> 1] =
                ((const unsigned*)sH)[base >> 1];
        }
    }
}

// d = relu(h2 @ Wd + bd); out = sigmoid(d @ Ws + bs).  h2: 64x256 bf16; Wd: 256x128 f32.
__global__ void head_kernel(const bf16* __restrict__ h2, const float* __restrict__ Wd,
                            const float* __restrict__ bd, const float* __restrict__ Ws,
                            const float* __restrict__ bs, float* __restrict__ out) {
    __shared__ float sd[BB * UU];
    int tid = threadIdx.x;  // 512
    for (int cell = tid; cell < BB * UU; cell += 512) {
        int b = cell >> 7, j = cell & 127;
        float acc = bd[j];
        for (int k = 0; k < 2 * UU; ++k)
            acc += bf2f(h2[b * (2 * UU) + k]) * Wd[(size_t)k * UU + j];
        sd[cell] = fmaxf(acc, 0.0f);
    }
    __syncthreads();
    if (tid < BB) {
        float acc = bs[0];
        for (int j = 0; j < UU; ++j) acc += sd[tid * UU + j] * Ws[j];
        out[tid] = 1.0f / (1.0f + expf(-acc));
    }
}

// ---------- launch ----------
extern "C" void kernel_launch(void* const* d_in, const int* in_sizes, int n_in,
                              void* d_out, int out_size, void* d_ws, size_t ws_size,
                              hipStream_t stream) {
    const float* x    = (const float*)d_in[0];
    const float* W1fi = (const float*)d_in[1];
    const float* W1fh = (const float*)d_in[2];
    const float* b1f  = (const float*)d_in[3];
    const float* W1bi = (const float*)d_in[4];
    const float* W1bh = (const float*)d_in[5];
    const float* b1b  = (const float*)d_in[6];
    const float* W2fi = (const float*)d_in[7];
    const float* W2fh = (const float*)d_in[8];
    const float* b2f  = (const float*)d_in[9];
    const float* W2bi = (const float*)d_in[10];
    const float* W2bh = (const float*)d_in[11];
    const float* b2b  = (const float*)d_in[12];
    const float* Wd   = (const float*)d_in[13];
    const float* bd   = (const float*)d_in[14];
    const float* Ws   = (const float*)d_in[15];
    const float* bs   = (const float*)d_in[16];
    float* out = (float*)d_out;

    char* ws = (char*)d_ws;
    size_t off = 0;
    auto carve = [&](size_t bytes) -> char* {
        char* p = ws + off;
        off = (off + bytes + 255) & ~(size_t)255;
        return p;
    };
    bf16*     xbf   = (bf16*)carve((size_t)BT * FF * 2);
    unsigned* w1fi  = (unsigned*)carve((size_t)FF * NG * 2);
    unsigned* w1bi  = (unsigned*)carve((size_t)FF * NG * 2);
    unsigned* w1fh  = (unsigned*)carve((size_t)UU * NG * 2);
    unsigned* w1bh  = (unsigned*)carve((size_t)UU * NG * 2);
    unsigned* w2fi  = (unsigned*)carve((size_t)2 * UU * NG * 2);
    unsigned* w2bi  = (unsigned*)carve((size_t)2 * UU * NG * 2);
    unsigned* w2fh  = (unsigned*)carve((size_t)UU * NG * 2);
    unsigned* w2bh  = (unsigned*)carve((size_t)UU * NG * 2);
    float* maskb    = (float*)carve((size_t)BT * 4);
    float* zxA      = (float*)carve((size_t)BT * NG * 4);   // layer1-f then layer2-f (time-major)
    float* zxBp     = (float*)carve((size_t)BT * NG * 4);   // layer1-b then layer2-b
    bf16* h1        = (bf16*)carve((size_t)BT * 2 * UU * 2);
    bf16* h2        = (bf16*)carve((size_t)BB * 2 * UU * 2);

    // fp32 -> bf16 for the A-side activations
    cvt_f32_bf16<<<(BT * FF + 255) / 256, 256, 0, stream>>>(x, xbf, BT * FF);

    // fp32 -> packed-bf16 WMMA B-layout for all 8 LSTM weight matrices
    auto pack = [&](const float* s, unsigned* d, int K, int N) {
        int total = (K >> 5) * (N >> 4) * 256;
        pack_weights<<<(total + 255) / 256, 256, 0, stream>>>(s, d, K, N);
    };
    pack(W1fi, w1fi, FF, NG);
    pack(W1bi, w1bi, FF, NG);
    pack(W1fh, w1fh, UU, NG);
    pack(W1bh, w1bh, UU, NG);
    pack(W2fi, w2fi, 2 * UU, NG);
    pack(W2bi, w2bi, 2 * UU, NG);
    pack(W2fh, w2fh, UU, NG);
    pack(W2bh, w2bh, UU, NG);

    mask_kernel<<<(BT + 255) / 256, 256, 0, stream>>>(x, maskb);

    const size_t smemScan = 311296;  // 304 KB

    // Layer 1 input projections: (32768 x 64) @ (64 x 512), output time-major
    {
        int waves = (BT / 32) * (NG / 32);
        int blocks = (waves + 7) / 8;
        gemm_bf16<<<blocks, 256, 0, stream>>>(xbf, w1fi, b1f, zxA, BT, NG, FF);
        gemm_bf16<<<blocks, 256, 0, stream>>>(xbf, w1bi, b1b, zxBp, BT, NG, FF);
    }
    lstm_scan<<<2, 512, smemScan, stream>>>(zxA, zxBp, w1fh, w1bh, maskb, h1, 1);

    // Layer 2 input projections: (32768 x 256) @ (256 x 512)
    {
        int waves = (BT / 32) * (NG / 32);
        int blocks = (waves + 7) / 8;
        gemm_bf16<<<blocks, 256, 0, stream>>>(h1, w2fi, b2f, zxA, BT, NG, 2 * UU);
        gemm_bf16<<<blocks, 256, 0, stream>>>(h1, w2bi, b2b, zxBp, BT, NG, 2 * UU);
    }
    lstm_scan<<<2, 512, smemScan, stream>>>(zxA, zxBp, w2fh, w2bh, maskb, h2, 0);

    head_kernel<<<1, 512, 0, stream>>>(h2, Wd, bd, Ws, bs, out);
}
